// NestedAdaptiveVectorQuantizer_17102559773011
// MI455X (gfx1250) — matrix operand order repaired and checked by reading, not compiled, not run on hardware
//
#include <hip/hip_runtime.h>
#include <hip/hip_bf16.h>

typedef __attribute__((ext_vector_type(16))) _Float16 v16h;
typedef __attribute__((ext_vector_type(8)))  float    v8f;
typedef __attribute__((ext_vector_type(4)))  unsigned v4u;
typedef __attribute__((ext_vector_type(8)))  int      v8i;
typedef __attribute__((ext_vector_type(4)))  int      v4i;

#define NVEC      65536      // 16*64*64 spatial vectors
#define DIM       128
#define HW        4096       // 64*64
#define NLEVELS   8
#define CB_TOTAL  510        // 2+4+...+256
#define TOT_ELEMS 8388608.0f // 16*128*64*64

// amdgpu-toolchain (clang-23 + therock headers) uses the 6-arg TDM builtin;
// ROCm 7.2 (clang-22) uses the 5-arg form. The TDM header ships only with the
// former, so use its presence to pick the arity.
#if __has_include(<hip/amd_detail/amd_gfx1250_TDM.h>)
#define TDM_SIX_ARGS 1
#else
#define TDM_SIX_ARGS 0
#endif

// ---------------------------------------------------------------------------
// Kernel 1: expand the nested codebooks (510 entries x 128), norms, f16 copy.
// ---------------------------------------------------------------------------
__global__ __launch_bounds__(256) void nvq_build_cb(
    const float* __restrict__ W, float* __restrict__ cb,
    float* __restrict__ norms, _Float16* __restrict__ cbh,
    float* __restrict__ loss)
{
    const int tid = threadIdx.x;
    if (tid < NLEVELS) loss[tid] = 0.0f;

    for (int L = 1; L <= NLEVELS; ++L) {
        int i0, i1;
        if (L == 1) { i0 = 0; i1 = 1; }
        else {
            int m = (L - 1) >> 1;
            int e = ((L & 1) == 0) ? 1 : 0;
            i0 = 2 * m + 2 + e;            // 2m+3+e-1
            i1 = i0 + 1;
        }
        const int nL      = 1 << L;
        const int off     = nL - 2;        // level-L entries start here
        const int offPrev = (nL >> 1) - 2;
        const int total   = nL * DIM;
        for (int t = tid; t < total; t += blockDim.x) {
            int ent = t >> 7, c = t & 127;
            float base = (L == 1) ? 0.0f : cb[(size_t)(offPrev + (ent >> 1)) * DIM + c];
            int row = (ent & 1) ? i1 : i0;
            cb[(size_t)(off + ent) * DIM + c] = base + W[(size_t)row * DIM + c];
        }
        __syncthreads();
    }

    for (int e = tid; e < CB_TOTAL; e += blockDim.x) {
        float s = 0.0f;
        for (int c = 0; c < DIM; ++c) {
            float v = cb[(size_t)e * DIM + c];
            s += v * v;
            cbh[(size_t)e * DIM + c] = (_Float16)v;
        }
        norms[e] = s;
    }
}

// ---------------------------------------------------------------------------
// Kernel 2: WMMA nearest-neighbor quantize. One wave handles 16 x-vectors.
// The whole f16 codebook is DMA'd once per workgroup into LDS by the TDM.
// ---------------------------------------------------------------------------
__global__ __launch_bounds__(256) void nvq_quantize(
    const float*    __restrict__ x,     // (16,128,64,64)
    const float*    __restrict__ cb,    // 510 x 128 f32
    const float*    __restrict__ norms, // 510
    const _Float16* __restrict__ cbh,   // 510 x 128 f16 (TDM source)
    float*          __restrict__ out,   // (8,16,128,64,64)
    float*          __restrict__ loss)  // 8 accumulators
{
    __shared__ __align__(32) _Float16 ldsCB[CB_TOTAL * DIM]; // 130,560 B
    __shared__ int   sIdx[8][16];
    __shared__ float sXn [8][16];

    const int lane = threadIdx.x & 31;
    const int wave = threadIdx.x >> 5;
    const int gw   = blockIdx.x * 8 + wave;
    const int r0   = gw << 4;                 // first of this wave's 16 rows
    const int m    = lane & 15;               // A row / B column / output row
    const int hi   = lane >> 4;               // lane half (K-group select)

    // ---- Wave 0 kicks off the TDM copy: global cbh -> LDS (1-D tile) ----
    if (threadIdx.x == 0) {
        const unsigned nelem  = CB_TOTAL * DIM;                 // 65280 f16
        const unsigned ldsoff = (unsigned)(uintptr_t)&ldsCB[0]; // flat[31:0] = LDS offset
        const unsigned long long ga = (unsigned long long)(uintptr_t)cbh;
        v4u g0 = { 1u,                                          // count=1 (valid), user mode
                   ldsoff,                                      // D#.lds_addr
                   (unsigned)(ga & 0xFFFFFFFFu),                // D#.global_addr[31:0]
                   (unsigned)((ga >> 32) & 0x01FFFFFFu) | (2u << 30) }; // [56:32] | type=2
        v8i g1 = { (int)(1u << 16),                             // data_size=1 (2 bytes)
                   (int)((nelem & 0xFFFFu) << 16),              // tensor_dim0[15:0]
                   (int)(((nelem >> 16) & 0xFFFFu) | (1u << 16)), // tensor_dim0[31:16] | tensor_dim1=1
                   (int)((nelem & 0xFFFFu) << 16),              // tensor_dim1[31:16]=0 | tile_dim0=nelem
                   0,                                           // tile_dim1=0, tile_dim2=0 (unused)
                   (int)nelem,                                  // tensor_dim0_stride[31:0]
                   0, 0 };
        v4i gz = { 0, 0, 0, 0 };
#if TDM_SIX_ARGS
        v8i gz8 = { 0, 0, 0, 0, 0, 0, 0, 0 };
        __builtin_amdgcn_tensor_load_to_lds(g0, g1, gz, gz, gz8, 0);
#else
        __builtin_amdgcn_tensor_load_to_lds(g0, g1, gz, gz, 0);
#endif
    }

    const int row = r0 + m;                   // flat spatial index (b,h,w)
    const int b   = row >> 12;
    const int hw  = row & (HW - 1);
    const float* xrow = x + (size_t)b * DIM * HW + hw;   // element c at [c*HW]

    __builtin_prefetch(xrow, 0, 1);           // global_prefetch_b8

    // ---- Pack A fragments (16x32 f16, contiguous k = 16*hi + s) + per-row |x|^2 ----
    v16h afrag[4];
    float xnp = 0.0f;
    #pragma unroll
    for (int kc = 0; kc < 4; ++kc) {
        const int kb = kc * 32 + hi * 16;
        #pragma unroll
        for (int j = 0; j < 8; ++j) {
            float v0 = xrow[(size_t)(kb + j) * HW];
            float v1 = xrow[(size_t)(kb + 8 + j) * HW];
            xnp += v0 * v0 + v1 * v1;
            afrag[kc][j]     = (_Float16)v0;   // K = kb + j
            afrag[kc][j + 8] = (_Float16)v1;   // K = kb + 8 + j
        }
    }
    float xn = xnp + __shfl_xor(xnp, 16, 32);  // full |x_m|^2 in lanes m and m+16
    if (hi == 0) sXn[wave][m] = xn;

    // TDM done for wave 0, then publish LDS codebook to all waves.
    if (threadIdx.x == 0) __builtin_amdgcn_s_wait_tensorcnt(0);
    __syncthreads();

    // ---- Per-level nearest-neighbor search ----
    for (int L = 1; L <= NLEVELS; ++L) {
        const int nL    = 1 << L;
        const int off   = nL - 2;
        const int ntile = (nL + 15) >> 4;

        float bestS[8];
        int   bestI[8];
        #pragma unroll
        for (int r = 0; r < 8; ++r) { bestS[r] = -3.0e38f; bestI[r] = 0; }

        for (int t = 0; t < ntile; ++t) {
            const int col  = t * 16 + m;
            const int colc = (col < nL) ? col : (nL - 1);
            const int ent  = off + colc;
            const _Float16* bp = &ldsCB[(unsigned)ent * DIM + hi * 16];

            v16h b0 = *(const v16h*)(bp +  0);   // K =   0..31 group (ds_load)
            v16h b1 = *(const v16h*)(bp + 32);   // K =  32..63
            v16h b2 = *(const v16h*)(bp + 64);   // K =  64..95
            v16h b3 = *(const v16h*)(bp + 96);   // K =  96..127

            v8f acc = {};
            acc = __builtin_amdgcn_wmma_f32_16x16x32_f16(false, afrag[0], false, b0, (short)0, acc, false, false);
            acc = __builtin_amdgcn_wmma_f32_16x16x32_f16(false, afrag[1], false, b1, (short)0, acc, false, false);
            acc = __builtin_amdgcn_wmma_f32_16x16x32_f16(false, afrag[2], false, b2, (short)0, acc, false, false);
            acc = __builtin_amdgcn_wmma_f32_16x16x32_f16(false, afrag[3], false, b3, (short)0, acc, false, false);

            const float nrm   = norms[ent];
            const bool  valid = (col < nL);
            #pragma unroll
            for (int r = 0; r < 8; ++r) {        // acc[r] = S[row r+8*hi][col]
                float s = valid ? (2.0f * acc[r] - nrm) : -3.0e38f;
                if (s > bestS[r]) { bestS[r] = s; bestI[r] = colc; }
            }
        }

        // argmax across the 16 lanes of each half (columns), lowest index on tie
        #pragma unroll
        for (int d = 1; d < 16; d <<= 1) {
            #pragma unroll
            for (int r = 0; r < 8; ++r) {
                float os = __shfl_xor(bestS[r], d, 32);
                int   oi = __shfl_xor(bestI[r], d, 32);
                if (os > bestS[r] || (os == bestS[r] && oi < bestI[r])) {
                    bestS[r] = os; bestI[r] = oi;
                }
            }
        }

        // lanes 0 / 16 publish indices + accumulate loss:  sum(|x|^2 - best_score)
        if (m == 0) {
            float ls = 0.0f;
            #pragma unroll
            for (int r = 0; r < 8; ++r) {
                sIdx[wave][hi * 8 + r] = bestI[r];
                ls += sXn[wave][hi * 8 + r] - bestS[r];
            }
            atomicAdd(&loss[L - 1], ls);
        }
        __syncthreads();

        // ---- Write q = cb[idx]: lane covers row m, channels [64*hi, 64*hi+64) ----
        const int idx = sIdx[wave][m];
        const float* qv = cb + (size_t)(off + idx) * DIM;
        float* obase = out + (((size_t)(L - 1) * 16 + b) * DIM) * HW + hw;
        const int c0 = hi * 64;
        #pragma unroll 8
        for (int c = c0; c < c0 + 64; ++c) {
            obase[(size_t)c * HW] = qv[c];
        }
        __syncthreads();   // sIdx reused next level
    }
}

// ---------------------------------------------------------------------------
// Kernel 3: scale losses:  (COEFS[L] + 0.4) * sum / numel
// ---------------------------------------------------------------------------
__global__ void nvq_finalize(const float* __restrict__ loss,
                             float* __restrict__ outLoss)
{
    const float coefs[NLEVELS] = {1.5f, 1.2f, 1.0f, 0.9f, 0.82f, 0.69f, 0.65f, 0.56f};
    int t = threadIdx.x;
    if (t < NLEVELS) {
        outLoss[t] = (coefs[t] + 0.4f) * loss[t] * (1.0f / TOT_ELEMS);
    }
}

// ---------------------------------------------------------------------------
extern "C" void kernel_launch(void* const* d_in, const int* in_sizes, int n_in,
                              void* d_out, int out_size, void* d_ws, size_t ws_size,
                              hipStream_t stream) {
    const float* x = (const float*)d_in[0];   // (16,128,64,64) f32
    const float* W = (const float*)d_in[1];   // (256,128) f32
    (void)in_sizes; (void)n_in; (void)out_size; (void)ws_size;

    char* ws = (char*)d_ws;
    float*    cb    = (float*)   (ws + 0);        // 510*128*4   = 261120 B
    float*    norms = (float*)   (ws + 261120);   // 510*4       =   2040 B
    float*    loss  = (float*)   (ws + 263168);   // 8*4 (32B-aligned)
    _Float16* cbh   = (_Float16*)(ws + 263424);   // 510*128*2 (256B-aligned)

    float* out     = (float*)d_out;               // (8,16,128,64,64)
    float* outLoss = out + 67108864ull;           // 8 losses appended

    hipLaunchKernelGGL(nvq_build_cb, dim3(1),   dim3(256), 0, stream, W, cb, norms, cbh, loss);
    hipLaunchKernelGGL(nvq_quantize, dim3(512), dim3(256), 0, stream, x, cb, norms, cbh, out, loss);
    hipLaunchKernelGGL(nvq_finalize, dim3(1),   dim3(32),  0, stream, loss, outLoss);
}
